// SelfAttention_55972013802215
// MI455X (gfx1250) — matrix operand order repaired and checked by reading, not compiled, not run on hardware
//
#include <hip/hip_runtime.h>
#include <hip/hip_bf16.h>

typedef __attribute__((ext_vector_type(16))) __bf16 bf16x16;
typedef __attribute__((ext_vector_type(8)))  __bf16 bf16x8;
typedef __attribute__((ext_vector_type(8)))  float  f32x8;
typedef __attribute__((ext_vector_type(4)))  float  f32x4;

union Frag16 { bf16x16 v; bf16x8 h[2]; };

constexpr int   kB  = 2;
constexpr int   kC  = 512;
constexpr int   kH  = 8;
constexpr int   kD  = 64;     // dim_head
constexpr int   kN  = 4096;   // sequence length (64*64)
constexpr int   kM1 = 3 * kC; // 1536 qkv output channels
constexpr float kScale = 0.125f; // dim_head^-0.5

// ---------------------------------------------------------------------------
// Kernel 1: qkv = qkv_w @ x + qkv_b. Q,K scattered to [b,h,n,d]; V scattered
// pre-transposed to [b,h,d,n] so the attention kernel needs no V transpose.
// Block tile 128(M) x 128(N), K-step 32, double-buffered LDS, bf16 WMMA.
// ---------------------------------------------------------------------------
__global__ __launch_bounds__(256) void qkv_gemm_kernel(
    const float* __restrict__ x,      // [B, C, N]
    const float* __restrict__ w,      // [3C, C]
    const float* __restrict__ bias,   // [3C]
    __bf16* __restrict__ Q,           // [B,H,N,D]
    __bf16* __restrict__ K,           // [B,H,N,D]
    __bf16* __restrict__ V)           // [B,H,D,N]  (transposed!)
{
    constexpr int BM = 128, BN = 128, BK = 32, LDA = BK + 8; // 80B rows (16B mult)
    __shared__ __align__(16) __bf16 As[2][BM][LDA];          // W tile  [m][k]
    __shared__ __align__(16) __bf16 Bs[2][BN][LDA];          // x tile^T [n][k]

    const int t    = threadIdx.x;
    const int lane = t & 31;
    const int wid  = t >> 5;        // 0..7
    const int wm   = wid >> 2;      // 0..1
    const int wn   = wid & 3;       // 0..3
    const int half = lane >> 4;
    const int l16  = lane & 15;

    const int n0 = blockIdx.x * BN;
    const int m0 = blockIdx.y * BM;
    const int b  = blockIdx.z;

    f32x8 acc[4][2] = {};

    // Batch all global loads first, then convert + ds_store (no per-chunk wait).
    auto stage = [&](int kt, int buf) {
        const int k0 = kt * BK;
        f32x4 fa[4], fb[4];
        #pragma unroll
        for (int p = 0; p < 4; ++p) {
            const int li = (t + p * 256) * 4;      // 0..4095, step 4
            const int row = li >> 5, col = li & 31;
            fa[p] = *(const f32x4*)(w + (size_t)(m0 + row) * kC + k0 + col);
            const int cl = li >> 7, nl = li & 127;
            fb[p] = *(const f32x4*)(x + ((size_t)b * kC + k0 + cl) * kN + n0 + nl);
        }
        #pragma unroll
        for (int p = 0; p < 4; ++p) {
            const int li = (t + p * 256) * 4;
            const int row = li >> 5, col = li & 31;
            As[buf][row][col + 0] = (__bf16)fa[p].x;
            As[buf][row][col + 1] = (__bf16)fa[p].y;
            As[buf][row][col + 2] = (__bf16)fa[p].z;
            As[buf][row][col + 3] = (__bf16)fa[p].w;
            const int cl = li >> 7, nl = li & 127;
            Bs[buf][nl + 0][cl] = (__bf16)fb[p].x;
            Bs[buf][nl + 1][cl] = (__bf16)fb[p].y;
            Bs[buf][nl + 2][cl] = (__bf16)fb[p].z;
            Bs[buf][nl + 3][cl] = (__bf16)fb[p].w;
        }
    };

    stage(0, 0);
    __syncthreads();
    #pragma unroll 1
    for (int kt = 0; kt < kC / BK; ++kt) {
        const int buf = kt & 1;
        if (kt + 1 < kC / BK) stage(kt + 1, buf ^ 1);
        if (kt + 2 < kC / BK) {      // pull k+2 tiles toward L2/L0
            const int k2 = (kt + 2) * BK;
            __builtin_prefetch(w + (size_t)(m0 + (t >> 1)) * kC + k2 + (t & 1) * 16, 0, 3);
            __builtin_prefetch(x + ((size_t)b * kC + k2 + (t >> 3)) * kN + n0 + (t & 7) * 16, 0, 3);
        }

        Frag16 a[4], bb[2];
        #pragma unroll
        for (int mt = 0; mt < 4; ++mt) {   // A frag: lane=row, K = 8*half+j / 16+8*half+j
            const __bf16* p = &As[buf][wm * 64 + mt * 16 + l16][8 * half];
            a[mt].h[0] = *(const bf16x8*)p;
            a[mt].h[1] = *(const bf16x8*)(p + 16);
        }
        #pragma unroll
        for (int nt = 0; nt < 2; ++nt) {   // B frag: lane=col, K = 8*half+j / 16+8*half+j
            const __bf16* p = &Bs[buf][wn * 32 + nt * 16 + l16][8 * half];
            bb[nt].h[0] = *(const bf16x8*)p;
            bb[nt].h[1] = *(const bf16x8*)(p + 16);
        }
        #pragma unroll
        for (int mt = 0; mt < 4; ++mt)
            #pragma unroll
            for (int nt = 0; nt < 2; ++nt)
                acc[mt][nt] = __builtin_amdgcn_wmma_f32_16x16x32_bf16(
                    false, a[mt].v, false, bb[nt].v, (short)0, acc[mt][nt], false, false);
        __syncthreads();
    }

    // Epilogue: bias, split q/k/v heads
    #pragma unroll
    for (int mt = 0; mt < 4; ++mt) {
        #pragma unroll
        for (int nt = 0; nt < 2; ++nt) {
            const int nn = n0 + wn * 32 + nt * 16 + l16;
            #pragma unroll
            for (int r = 0; r < 8; ++r) {
                const int o = m0 + wm * 64 + mt * 16 + r + 8 * half;
                const float vo = acc[mt][nt][r] + bias[o];
                const int g = o >> 9;          // 0=q 1=k 2=v
                const int c = o & 511;
                const int hh = c >> 6, d = c & 63;
                if (g == 2) {                  // V transposed: [b,h,d,n]
                    V[(((size_t)b * kH + hh) * kD + d) * kN + nn] = (__bf16)vo;
                } else {
                    __bf16* dst = (g == 0) ? Q : K;
                    dst[(((size_t)b * kH + hh) * kN + nn) * kD + d] = (__bf16)vo;
                }
            }
        }
    }
}

// ---------------------------------------------------------------------------
// Kernel 2: flash attention, barrier-free. Block = 128 query rows of one
// (b,h); wave = 16 rows. K rows ([n,d]) and V^T rows ([d,n]) are already in
// B-fragment order -> direct 16B global loads (two per fragment), no LDS transposes.
// Only per-wave LDS relayout of P (D-layout -> A-layout) remains.
// ---------------------------------------------------------------------------
__global__ __launch_bounds__(256) void attn_kernel(
    const __bf16* __restrict__ Q,    // [B,H,N,D]
    const __bf16* __restrict__ K,    // [B,H,N,D]
    const __bf16* __restrict__ V,    // [B,H,D,N]
    __bf16* __restrict__ O)          // [B,H,N,D]
{
    constexpr int BQ = 128, BKV = 64, LDP = kD + 8;   // 144B rows
    __shared__ __align__(16) __bf16 Pl[8][16][LDP];   // per-wave P (A-layout source)

    const int t = threadIdx.x, lane = t & 31, w = t >> 5;
    const int half = lane >> 4, l16 = lane & 15;
    const int q0 = blockIdx.x * BQ;
    const int h  = blockIdx.y, b = blockIdx.z;

    const size_t hb = ((size_t)b * kH + h) * kN;
    const __bf16* Qg = Q + hb * kD;
    const __bf16* Kg = K + hb * kD;
    const __bf16* Vg = V + hb * kD;   // == V + ((b*H+h)*D)*N

    // Q fragments: fixed for the whole block (A of S = Q K^T)
    Frag16 aq[2];
    {
        const __bf16* qp = Qg + (size_t)(q0 + w * 16 + l16) * kD;
        #pragma unroll
        for (int ks = 0; ks < 2; ++ks) {
            aq[ks].h[0] = *(const bf16x8*)(qp + ks * 32 + 8 * half);
            aq[ks].h[1] = *(const bf16x8*)(qp + ks * 32 + 8 * half + 16);
        }
    }

    f32x8 oacc[4] = {};
    float mst[8], lst[8];
    #pragma unroll
    for (int r = 0; r < 8; ++r) { mst[r] = -1e30f; lst[r] = 0.f; }

    #pragma unroll 1
    for (int kv0 = 0; kv0 < kN; kv0 += BKV) {
        if (kv0 + BKV < kN) {   // prefetch next K rows and V^T columns
            const int kvn = kv0 + BKV;
            __builtin_prefetch(Kg + (size_t)(kvn + (t >> 2)) * kD + (t & 3) * 16, 0, 3);
            __builtin_prefetch(Vg + (size_t)(t >> 2) * kN + kvn + (t & 3) * 16, 0, 3);
        }

        // S = Q K^T : direct global B-fragments from K rows
        f32x8 sacc[4] = {};
        #pragma unroll
        for (int nt = 0; nt < 4; ++nt) {
            const __bf16* kp = Kg + (size_t)(kv0 + nt * 16 + l16) * kD + 8 * half;
            #pragma unroll
            for (int ks = 0; ks < 2; ++ks) {
                Frag16 bk;
                bk.h[0] = *(const bf16x8*)(kp + ks * 32);
                bk.h[1] = *(const bf16x8*)(kp + ks * 32 + 16);
                sacc[nt] = __builtin_amdgcn_wmma_f32_16x16x32_bf16(
                    false, aq[ks].v, false, bk.v, (short)0, sacc[nt], false, false);
            }
        }

        // Online softmax: row r+8*half lives in VGPR r across a 16-lane half.
        #pragma unroll
        for (int r = 0; r < 8; ++r) {
            float bm = -1e30f;
            #pragma unroll
            for (int nt = 0; nt < 4; ++nt) {
                sacc[nt][r] *= kScale;
                bm = fmaxf(bm, sacc[nt][r]);
            }
            #pragma unroll
            for (int m = 1; m < 16; m <<= 1) bm = fmaxf(bm, __shfl_xor(bm, m, 16));
            const float mnew = fmaxf(mst[r], bm);
            const float corr = __expf(mst[r] - mnew);
            mst[r] = mnew;
            float psum = 0.f;
            #pragma unroll
            for (int nt = 0; nt < 4; ++nt) {
                const float pe = __expf(sacc[nt][r] - mnew);
                psum += pe;
                Pl[w][r + 8 * half][nt * 16 + l16] = (__bf16)pe;
            }
            #pragma unroll
            for (int m = 1; m < 16; m <<= 1) psum += __shfl_xor(psum, m, 16);
            lst[r] = lst[r] * corr + psum;
            #pragma unroll
            for (int dt = 0; dt < 4; ++dt) oacc[dt][r] *= corr;
        }

        // O += P V : A = P via per-wave LDS relayout (same-wave LDS is
        // in-order, no barrier), B = V^T rows straight from global.
        Frag16 ap[2];
        #pragma unroll
        for (int ks = 0; ks < 2; ++ks) {
            const __bf16* pp = &Pl[w][l16][ks * 32 + 8 * half];
            ap[ks].h[0] = *(const bf16x8*)pp;
            ap[ks].h[1] = *(const bf16x8*)(pp + 16);
        }
        #pragma unroll
        for (int dt = 0; dt < 4; ++dt) {
            const __bf16* vp = Vg + (size_t)(dt * 16 + l16) * kN + kv0 + 8 * half;
            #pragma unroll
            for (int ks = 0; ks < 2; ++ks) {
                Frag16 bv;
                bv.h[0] = *(const bf16x8*)(vp + ks * 32);
                bv.h[1] = *(const bf16x8*)(vp + ks * 32 + 16);
                oacc[dt] = __builtin_amdgcn_wmma_f32_16x16x32_bf16(
                    false, ap[ks].v, false, bv.v, (short)0, oacc[dt], false, false);
            }
        }
    }

    // Normalize and store O in [b,h,n,d] bf16
    #pragma unroll
    for (int dt = 0; dt < 4; ++dt)
        #pragma unroll
        for (int r = 0; r < 8; ++r) {
            const float vo = oacc[dt][r] / lst[r];
            O[(hb + q0 + w * 16 + r + 8 * half) * kD + dt * 16 + l16] = (__bf16)vo;
        }
}

// ---------------------------------------------------------------------------
// Kernel 3: out = proj_w @ attn_out + proj_b, fp32 output. B tile is bf16 and
// layout-preserving -> staged with CDNA5 async global->LDS copies (ASYNCcnt),
// drained with s_wait_asynccnt before the tile barrier.
// ---------------------------------------------------------------------------
__global__ __launch_bounds__(256) void proj_gemm_kernel(
    const __bf16* __restrict__ Ob,    // [B,H,N,D]  == [b, c, n] with c=h*64+d
    const float* __restrict__ w,      // [C, C]
    const float* __restrict__ bias,   // [C]
    float* __restrict__ out)          // [B, C, N]
{
    constexpr int BM = 128, BN = 128, BK = 32, LDA = BK + 8;
    __shared__ __align__(16) __bf16 As[2][BM][LDA];
    __shared__ __align__(16) __bf16 Bs[2][BN][LDA];

    const int t = threadIdx.x, lane = t & 31, wid = t >> 5;
    const int wm = wid >> 2, wn = wid & 3;
    const int half = lane >> 4, l16 = lane & 15;
    const int n0 = blockIdx.x * BN, m0 = blockIdx.y * BM, b = blockIdx.z;

    f32x8 acc[4][2] = {};

    auto stage = [&](int kt, int buf) {
        const int k0 = kt * BK;
        // B: c-tile of 32 stays inside one head -> contiguous 16B chunks.
        // Async copy global -> LDS, no VGPR round-trip (ASYNCcnt-tracked).
        const int hh = k0 >> 6, d0 = k0 & 63;
        #pragma unroll
        for (int p = 0; p < 2; ++p) {
            const int chunk = t + p * 256;               // 0..511
            const int nl = chunk >> 2, j0 = (chunk & 3) * 8;
            const unsigned long long gaddr = (unsigned long long)
                (Ob + (((size_t)b * kH + hh) * kN + n0 + nl) * kD + d0 + j0);
            const unsigned laddr = (unsigned)(unsigned long long)&Bs[buf][nl][j0];
            asm volatile("global_load_async_to_lds_b128 %0, %1, off"
                         :: "v"(laddr), "v"(gaddr) : "memory");
        }
        // A: proj_w fp32 -> bf16 (must pass through VGPRs for conversion)
        f32x4 fa[4];
        #pragma unroll
        for (int p = 0; p < 4; ++p) {
            const int li = (t + p * 256) * 4;
            const int row = li >> 5, col = li & 31;
            fa[p] = *(const f32x4*)(w + (size_t)(m0 + row) * kC + k0 + col);
        }
        #pragma unroll
        for (int p = 0; p < 4; ++p) {
            const int li = (t + p * 256) * 4;
            const int row = li >> 5, col = li & 31;
            As[buf][row][col + 0] = (__bf16)fa[p].x;
            As[buf][row][col + 1] = (__bf16)fa[p].y;
            As[buf][row][col + 2] = (__bf16)fa[p].z;
            As[buf][row][col + 3] = (__bf16)fa[p].w;
        }
    };

    stage(0, 0);
    asm volatile("s_wait_asynccnt 0" ::: "memory");
    __syncthreads();
    #pragma unroll 1
    for (int kt = 0; kt < kC / BK; ++kt) {
        const int buf = kt & 1;
        if (kt + 1 < kC / BK) stage(kt + 1, buf ^ 1);
        if (kt + 2 < kC / BK) {
            const int k2 = (kt + 2) * BK;
            __builtin_prefetch(w + (size_t)(m0 + (t >> 1)) * kC + k2 + (t & 1) * 16, 0, 3);
        }

        Frag16 a[4], bb[2];
        #pragma unroll
        for (int mt = 0; mt < 4; ++mt) {
            const __bf16* p = &As[buf][wm * 64 + mt * 16 + l16][8 * half];
            a[mt].h[0] = *(const bf16x8*)p;
            a[mt].h[1] = *(const bf16x8*)(p + 16);
        }
        #pragma unroll
        for (int nt = 0; nt < 2; ++nt) {
            const __bf16* p = &Bs[buf][wn * 32 + nt * 16 + l16][8 * half];
            bb[nt].h[0] = *(const bf16x8*)p;
            bb[nt].h[1] = *(const bf16x8*)(p + 16);
        }
        #pragma unroll
        for (int mt = 0; mt < 4; ++mt)
            #pragma unroll
            for (int nt = 0; nt < 2; ++nt)
                acc[mt][nt] = __builtin_amdgcn_wmma_f32_16x16x32_bf16(
                    false, a[mt].v, false, bb[nt].v, (short)0, acc[mt][nt], false, false);

        asm volatile("s_wait_asynccnt 0" ::: "memory");  // next tile's B landed
        __syncthreads();
    }

    #pragma unroll
    for (int mt = 0; mt < 4; ++mt)
        #pragma unroll
        for (int nt = 0; nt < 2; ++nt) {
            const int nn = n0 + wn * 32 + nt * 16 + l16;
            #pragma unroll
            for (int r = 0; r < 8; ++r) {
                const int o = m0 + wm * 64 + mt * 16 + r + 8 * half;
                out[((size_t)b * kC + o) * kN + nn] = acc[mt][nt][r] + bias[o];
            }
        }
}

// ---------------------------------------------------------------------------
extern "C" void kernel_launch(void* const* d_in, const int* in_sizes, int n_in,
                              void* d_out, int out_size, void* d_ws, size_t ws_size,
                              hipStream_t stream) {
    (void)in_sizes; (void)n_in; (void)out_size; (void)ws_size;
    const float* x      = (const float*)d_in[0];
    const float* qkv_w  = (const float*)d_in[1];
    const float* qkv_b  = (const float*)d_in[2];
    const float* proj_w = (const float*)d_in[3];
    const float* proj_b = (const float*)d_in[4];
    float* out = (float*)d_out;

    const size_t per = (size_t)kB * kH * kN * kD;   // 4Mi bf16 elems each
    __bf16* Qw = (__bf16*)d_ws;
    __bf16* Kw = Qw + per;
    __bf16* Vw = Kw + per;   // stored transposed [b,h,d,n]
    __bf16* Ow = Vw + per;   // total 32 MB of workspace

    qkv_gemm_kernel<<<dim3(kN / 128, kM1 / 128, kB), 256, 0, stream>>>(
        x, qkv_w, qkv_b, Qw, Kw, Vw);
    attn_kernel<<<dim3(kN / 128, kH, kB), 256, 0, stream>>>(Qw, Kw, Vw, Ow);
    proj_gemm_kernel<<<dim3(kN / 128, kC / 128, kB), 256, 0, stream>>>(
        Ow, proj_w, proj_b, out);
}
